// CrossAttention_15470472200717
// MI455X (gfx1250) — matrix operand order repaired and checked by reading, not compiled, not run on hardware
//
#include <hip/hip_runtime.h>
#include <hip/hip_bf16.h>
#include <math.h>

typedef __attribute__((ext_vector_type(16))) _Float16 v16h;
typedef __attribute__((ext_vector_type(8)))  _Float16 v8h;
typedef __attribute__((ext_vector_type(8)))  float    v8f;
typedef __attribute__((ext_vector_type(4)))  unsigned int v4u;
typedef __attribute__((ext_vector_type(8)))  int      v8i;
typedef __attribute__((ext_vector_type(4)))  int      v4i;

#define B_N   4
#define C_DIM 256
#define L_SEQ 4096
#define HD    128
#define BK    32

#if defined(__HIP_DEVICE_COMPILE__) && defined(__has_builtin)
#if __has_builtin(__builtin_amdgcn_tensor_load_to_lds)
#define HAVE_TDM 1
#endif
#endif
#ifndef HAVE_TDM
#define HAVE_TDM 0
#endif

__device__ __forceinline__ v8f wmma_f16(v16h a, v16h b, v8f c) {
  // D = A(16x32) * B(32x16) + C, f16 in / f32 acc
  return __builtin_amdgcn_wmma_f32_16x16x32_f16(false, a, false, b, (short)0, c,
                                                false, false);
}

__device__ __forceinline__ v16h cat8(v8h lo, v8h hi) {
  return __builtin_shufflevector(lo, hi, 0, 1, 2, 3, 4, 5, 6, 7,
                                 8, 9, 10, 11, 12, 13, 14, 15);
}

#if HAVE_TDM
// Issue a TDM 2D tile load (D# per CDNA5 ISA ch.8): data_size=2B, type=image.
// 1D transfers: pass tile_d1 = 0.
__device__ __forceinline__ void tdm_load_2d(
    unsigned int lds_off, unsigned long long gaddr, unsigned int tensor_d0,
    unsigned int tensor_d1, unsigned int tile_d0, unsigned int tile_d1,
    unsigned long long stride0) {
  v4u g0;
  g0[0] = 1u;  // count=1, is_restore=0, gather_mode=0
  g0[1] = lds_off;
  g0[2] = (unsigned int)(gaddr & 0xffffffffull);
  g0[3] = (unsigned int)((gaddr >> 32) & 0x01ffffffull) | (2u << 30);  // type=2
  v8i g1;
  g1[0] = (int)(1u << 16);  // workgroup_mask=0, data_size=1 (2 bytes)
  g1[1] = (int)((tensor_d0 & 0xffffu) << 16);  // abar_addr=0 | dim0[15:0]
  g1[2] = (int)((tensor_d0 >> 16) | ((tensor_d1 & 0xffffu) << 16));
  g1[3] = (int)((tensor_d1 >> 16) | (tile_d0 << 16));
  g1[4] = (int)(tile_d1 & 0xffffu);  // tile_dim1 | tile_dim2=0
  g1[5] = (int)(stride0 & 0xffffffffull);
  g1[6] = (int)((stride0 >> 32) & 0xffffull);  // | dim1_stride[15:0]=0
  g1[7] = 0;
  v4i z4 = {0, 0, 0, 0};
#if __clang_major__ >= 23
  v8i z8 = {0, 0, 0, 0, 0, 0, 0, 0};
  __builtin_amdgcn_tensor_load_to_lds(g0, g1, z4, z4, z8, 0);
#else
  __builtin_amdgcn_tensor_load_to_lds(g0, g1, z4, z4, 0);
#endif
}
#endif

// ---------------------------------------------------------------------------
// f32 -> f16 elementwise cast (produces V^T directly: inputs are [B][C][L])
// ---------------------------------------------------------------------------
__global__ __launch_bounds__(256) void cast_f32_f16_kernel(
    const float* __restrict__ in, _Float16* __restrict__ out, int n) {
  int i = blockIdx.x * blockDim.x + threadIdx.x;
  if (i < n) out[i] = (_Float16)in[i];
}

// ---------------------------------------------------------------------------
// Projection: Out[l][d] = (sum_c X[c][l] * W[d][c]) * scale[d] + shift[d]
// mode 0: BN eval  (p0=gamma p1=beta p2=mean p3=var); mode 1: bias (p0)
// ---------------------------------------------------------------------------
__global__ __launch_bounds__(256) void proj_gemm_kernel(
    const float* __restrict__ Xall,   // [B][C][L]
    const float* __restrict__ W,      // [HD][C]
    const float* __restrict__ p0, const float* __restrict__ p1,
    const float* __restrict__ p2, const float* __restrict__ p3,
    int mode,
    _Float16* __restrict__ Outall) {  // [B][L][HD]
  __shared__ _Float16 sX[128][BK];    // transposed input tile, 8 KB

  const int b = blockIdx.y;
  const int ltile = blockIdx.x;
  const int tid = threadIdx.x;
  const int wave = tid >> 5, lane = tid & 31;
  const int ln16 = lane & 15, grp = lane >> 4;

  const float* Xb = Xall + (size_t)b * C_DIM * L_SEQ;
  _Float16* Ob = Outall + (size_t)b * L_SEQ * HD;

  v8f acc[8];
  for (int i = 0; i < 8; ++i)
    for (int e = 0; e < 8; ++e) acc[i][e] = 0.f;

  const int cl = tid >> 3;        // 0..31 : c row within chunk
  const int lo = (tid & 7) * 16;  // 0..112: l offset

  for (int cc = 0; cc < C_DIM / BK; ++cc) {
    __syncthreads();
    const float* xp = Xb + (size_t)(cc * BK + cl) * L_SEQ + ltile * 128 + lo;
#pragma unroll
    for (int i = 0; i < 16; ++i) sX[lo + i][cl] = (_Float16)xp[i];
    __syncthreads();

    const _Float16* ap = &sX[wave * 16 + ln16][grp * 16];
    v16h a = cat8(*(const v8h*)ap, *(const v8h*)(ap + 8));

#pragma unroll
    for (int dc = 0; dc < 8; ++dc) {
      const float* wp = W + (size_t)(dc * 16 + ln16) * C_DIM + cc * BK + grp * 16;
      v16h bf;
#pragma unroll
      for (int e = 0; e < 16; ++e) bf[e] = (_Float16)wp[e];
      acc[dc] = wmma_f16(a, bf, acc[dc]);
    }
  }

#pragma unroll
  for (int dc = 0; dc < 8; ++dc) {
    const int d = dc * 16 + ln16;
    float s, t;
    if (mode == 0) {
      float inv = rsqrtf(p3[d] + 1e-5f);
      s = p0[d] * inv;
      t = p1[d] - p2[d] * s;
    } else {
      s = 1.f;
      t = p0[d];
    }
#pragma unroll
    for (int v = 0; v < 8; ++v) {
      int l = ltile * 128 + wave * 16 + v + 8 * grp;
      Ob[(size_t)l * HD + d] = (_Float16)(acc[dc][v] * s + t);
    }
  }
}

// ---------------------------------------------------------------------------
// Flash attention: O = softmax(Q K^T * scale) V,  Hd=128, V width C=256.
// Q,K: [B][L][HD] f16.  VT: [B][C][L] f16.  O: [B][L][C] f32.
// 8 waves x 16 q-rows; 32-key blocks; K/V tiles double-buffered in LDS and
// fed by the Tensor Data Mover (wave 0 issues, TENSORcnt pipelines).
// ---------------------------------------------------------------------------
__global__ __launch_bounds__(256) void flash_attn_kernel(
    const _Float16* __restrict__ Qall, const _Float16* __restrict__ Kall,
    const _Float16* __restrict__ VTall, float* __restrict__ Oall,
    float score_scale) {
  __shared__ _Float16 sK[2][BK][HD];      // 2 x 8 KB
  __shared__ _Float16 sVT[2][C_DIM][BK];  // 2 x 16 KB

  const int b = blockIdx.y;
  const int qtile = blockIdx.x;
  const int tid = threadIdx.x;
  const int wave = tid >> 5, lane = tid & 31;
  const int ln16 = lane & 15, grp = lane >> 4;

  const _Float16* Qb = Qall + (size_t)b * L_SEQ * HD;
  const _Float16* Kb = Kall + (size_t)b * L_SEQ * HD;
  const _Float16* Vb = VTall + (size_t)b * C_DIM * L_SEQ;
  float* Ob = Oall + (size_t)b * L_SEQ * C_DIM;

  const int qrow0 = qtile * 128 + wave * 16;
  const int NB = L_SEQ / BK;

  // Preload Q^T B-frags: lane col = qrow, elems = dims (4 chunks of 32)
  v16h qb[4];
  {
    const _Float16* qp = Qb + (size_t)(qrow0 + ln16) * HD;
#pragma unroll
    for (int ch = 0; ch < 4; ++ch)
      qb[ch] = cat8(*(const v8h*)(qp + ch * 32 + grp * 8), *(const v8h*)(qp + ch * 32 + grp * 8 + 16));
  }

  v8f acc[16];  // O strip 16 x 256 (row = v+8*grp, col = c*16+ln16)
  for (int i = 0; i < 16; ++i)
    for (int e = 0; e < 8; ++e) acc[i][e] = 0.f;
  float m_row = -3.0e38f;
  float l_row = 0.f;

#if HAVE_TDM
  const bool tdm_wave = (wave == 0);
  unsigned int kLds[2], vLds[2];
  kLds[0] = (unsigned int)(size_t)&sK[0][0][0];
  kLds[1] = (unsigned int)(size_t)&sK[1][0][0];
  vLds[0] = (unsigned int)(size_t)&sVT[0][0][0];
  vLds[1] = (unsigned int)(size_t)&sVT[1][0][0];
  if (tdm_wave) {
    // tile 0 into buffer 0: K = 1D contiguous 4096 halfs, VT = 2D 32x256
    tdm_load_2d(kLds[0], (unsigned long long)(size_t)Kb, (unsigned)(L_SEQ * HD),
                0u, (unsigned)(BK * HD), 0u, (unsigned long long)(L_SEQ * HD));
    tdm_load_2d(vLds[0], (unsigned long long)(size_t)Vb, (unsigned)L_SEQ,
                (unsigned)C_DIM, (unsigned)BK, (unsigned)C_DIM,
                (unsigned long long)L_SEQ);
  }
#endif

  for (int kb = 0; kb < NB; ++kb) {
    const int cur = kb & 1;
#if HAVE_TDM
    if (tdm_wave) {
      if (kb + 1 < NB) {
        const size_t nkb = (size_t)(kb + 1);
        tdm_load_2d(kLds[cur ^ 1],
                    (unsigned long long)(size_t)(Kb + nkb * BK * HD),
                    (unsigned)(L_SEQ * HD), 0u, (unsigned)(BK * HD), 0u,
                    (unsigned long long)(L_SEQ * HD));
        tdm_load_2d(vLds[cur ^ 1],
                    (unsigned long long)(size_t)(Vb + nkb * BK),
                    (unsigned)L_SEQ, (unsigned)C_DIM, (unsigned)BK,
                    (unsigned)C_DIM, (unsigned long long)L_SEQ);
        __builtin_amdgcn_s_wait_tensorcnt((short)2);  // current tile landed
      } else {
        __builtin_amdgcn_s_wait_tensorcnt((short)0);
      }
    }
    __syncthreads();
#else
    __syncthreads();
    {
      const _Float16* ks = Kb + (size_t)kb * BK * HD + tid * 16;
      *(v16h*)(&sK[cur][0][0] + tid * 16) = *(const v16h*)ks;
      const _Float16* vs = Vb + (size_t)tid * L_SEQ + kb * BK;
      *(v16h*)(&sVT[cur][tid][0]) = *(const v16h*)vs;
      *(v16h*)(&sVT[cur][tid][0] + 16) = *(const v16h*)(vs + 16);
      if (kb + 1 < NB) {
        __builtin_prefetch(ks + BK * HD, 0, 1);
        __builtin_prefetch(vs + BK, 0, 1);
      }
    }
    __syncthreads();
#endif

    // S^T = Ktile(32x128) * Q^T(128x16); s0: keys 0-15 (M), s1: keys 16-31
    v8f s0, s1;
    for (int e = 0; e < 8; ++e) { s0[e] = 0.f; s1[e] = 0.f; }
#pragma unroll
    for (int ch = 0; ch < 4; ++ch) {
      const _Float16* k0 = &sK[cur][ln16][ch * 32 + grp * 8];
      const _Float16* k1 = &sK[cur][16 + ln16][ch * 32 + grp * 8];
      v16h a0 = cat8(*(const v8h*)k0, *(const v8h*)(k0 + 16));
      v16h a1 = cat8(*(const v8h*)k1, *(const v8h*)(k1 + 16));
      s0 = wmma_f16(a0, qb[ch], s0);
      s1 = wmma_f16(a1, qb[ch], s1);
    }

    // online softmax over keys for qrow = ln16 (16 keys in-lane + partner)
    float mt = -3.0e38f;
#pragma unroll
    for (int v = 0; v < 8; ++v) {
      s0[v] *= score_scale;
      s1[v] *= score_scale;
      mt = fmaxf(mt, fmaxf(s0[v], s1[v]));
    }
    mt = fmaxf(mt, __shfl_xor(mt, 16, 32));
    float m_new = fmaxf(m_row, mt);
    float corr = __expf(m_row - m_new);

    float p0[8], p1[8];
    float rs = 0.f;
#pragma unroll
    for (int v = 0; v < 8; ++v) {
      p0[v] = __expf(s0[v] - m_new);  // key = v + 8*grp
      p1[v] = __expf(s1[v] - m_new);  // key = 16 + v + 8*grp
      rs += p0[v] + p1[v];
    }
    rs += __shfl_xor(rs, 16, 32);
    l_row = l_row * corr + rs;
    m_row = m_new;

    // rescale O accumulators (O row = v + 8*grp; fetch that row's corr)
#pragma unroll
    for (int v = 0; v < 8; ++v) {
      float cf = __shfl(corr, v + 8 * grp, 32);
#pragma unroll
      for (int c = 0; c < 16; ++c) acc[c][v] *= cf;
    }

    // P^T registers == WMMA A-frag of P (16 qrows x 32 keys): pack in-lane
    v16h pa;
#pragma unroll
    for (int e = 0; e < 8; ++e) {
      pa[e] = (_Float16)p0[e];
      pa[e + 8] = (_Float16)p1[e];
    }

    // O += P * Vtile : 16 column chunks of C
#pragma unroll
    for (int c = 0; c < 16; ++c) {
      v16h vb = cat8(*(const v8h*)(&sVT[cur][c * 16 + ln16][grp * 8]), *(const v8h*)(&sVT[cur][c * 16 + ln16][grp * 8 + 16]));
      acc[c] = wmma_f16(pa, vb, acc[c]);
    }
    __syncthreads();  // compute done before DMA overwrites this buffer
  }

  // epilogue: divide by row sum, store f32
#pragma unroll
  for (int v = 0; v < 8; ++v) {
    float lr = __shfl(l_row, v + 8 * grp, 32);
    float inv = 1.f / lr;
    const int row = qrow0 + v + 8 * grp;
#pragma unroll
    for (int c = 0; c < 16; ++c)
      Ob[(size_t)row * C_DIM + c * 16 + ln16] = acc[c][v] * inv;
  }
}

// ---------------------------------------------------------------------------
extern "C" void kernel_launch(void* const* d_in, const int* in_sizes, int n_in,
                              void* d_out, int out_size, void* d_ws,
                              size_t ws_size, hipStream_t stream) {
  const float* input_i = (const float*)d_in[0];   // [B][C][H*W]
  const float* input_t = (const float*)d_in[1];   // [B][C][Lt]
  const float* w_img_q = (const float*)d_in[2];
  const float* bnq_g = (const float*)d_in[3];
  const float* bnq_b = (const float*)d_in[4];
  const float* bnq_m = (const float*)d_in[5];
  const float* bnq_v = (const float*)d_in[6];
  const float* w_img_k = (const float*)d_in[7];
  const float* bnk_g = (const float*)d_in[8];
  const float* bnk_b = (const float*)d_in[9];
  const float* bnk_m = (const float*)d_in[10];
  const float* bnk_v = (const float*)d_in[11];
  const float* w_text_k = (const float*)d_in[12];
  const float* b_text_k = (const float*)d_in[13];
  const float* w_text_q = (const float*)d_in[14];
  const float* b_text_q = (const float*)d_in[15];
  float* out = (float*)d_out;

  const size_t qk_elems = (size_t)B_N * L_SEQ * HD;    // 2 Mi halfs each
  const size_t v_elems = (size_t)B_N * C_DIM * L_SEQ;  // 4 Mi halfs each
  if (ws_size < (4 * qk_elems + 2 * v_elems) * sizeof(_Float16)) return;

  _Float16* q_img = (_Float16*)d_ws;
  _Float16* k_img = q_img + qk_elems;
  _Float16* t_q = k_img + qk_elems;
  _Float16* t_k = t_q + qk_elems;
  _Float16* vT_text = t_k + qk_elems;   // input_t cast -> V^T for pass 1
  _Float16* vT_img = vT_text + v_elems; // input_i cast -> V^T for pass 2

  const int nv = (int)v_elems;
  cast_f32_f16_kernel<<<(nv + 255) / 256, 256, 0, stream>>>(input_t, vT_text, nv);
  cast_f32_f16_kernel<<<(nv + 255) / 256, 256, 0, stream>>>(input_i, vT_img, nv);

  dim3 pgrid(L_SEQ / 128, B_N);
  proj_gemm_kernel<<<pgrid, 256, 0, stream>>>(input_i, w_img_q, bnq_g, bnq_b,
                                              bnq_m, bnq_v, 0, q_img);
  proj_gemm_kernel<<<pgrid, 256, 0, stream>>>(input_i, w_img_k, bnk_g, bnk_b,
                                              bnk_m, bnk_v, 0, k_img);
  proj_gemm_kernel<<<pgrid, 256, 0, stream>>>(input_t, w_text_k, b_text_k,
                                              nullptr, nullptr, nullptr, 1, t_k);
  proj_gemm_kernel<<<pgrid, 256, 0, stream>>>(input_t, w_text_q, b_text_q,
                                              nullptr, nullptr, nullptr, 1, t_q);

  const float scale = 11.313708498984761f;  // sqrt(Hd)
  dim3 agrid(L_SEQ / 128, B_N);
  // pass 1: output_i = softmax(i_q t_k^T) @ t_flat
  flash_attn_kernel<<<agrid, 256, 0, stream>>>(q_img, t_k, vT_text, out, scale);
  // pass 2: output_t = softmax(t_q i_k^T) @ i_v
  flash_attn_kernel<<<agrid, 256, 0, stream>>>(t_q, k_img, vT_img,
                                               out + (size_t)B_N * L_SEQ * C_DIM,
                                               scale);
}
